// DeepFM_13537736917033
// MI455X (gfx1250) — matrix operand order repaired and checked by reading, not compiled, not run on hardware
//
#include <hip/hip_runtime.h>
#include <hip/hip_bf16.h>
#include <math.h>

typedef __attribute__((ext_vector_type(16))) _Float16 v16h;
typedef __attribute__((ext_vector_type(8)))  float    v8f;

#define N_GENRES 18
#define EMB_OFFSET 38167
#define EMB_D 16
#define NBLOCKS 512

union AFrag { uint4 q[2]; v16h v; };
union PK { _Float16 h[2]; unsigned int u; };

__device__ __forceinline__ unsigned int pack2(float a, float b) {
    PK p; p.h[0] = (_Float16)a; p.h[1] = (_Float16)b; return p.u;
}
__device__ __forceinline__ float lo16(unsigned int u) { PK p; p.u = u; return (float)p.h[0]; }
__device__ __forceinline__ float hi16(unsigned int u) { PK p; p.u = u; return (float)p.h[1]; }

// Build a 16x32 f16 A-fragment from an LDS row of f16 (row_dw = dword pointer to
// the start of the row for this tile's M index). Layout per CDNA5 ISA 7.12.2:
// lane half hl=lane>>4: dwords [4*hl+0..3] and [4*hl+8..11] of the 32-K window.
__device__ __forceinline__ v16h load_afrag(const unsigned int* row_dw, int hl) {
    AFrag f;
    f.q[0] = *(const uint4*)(row_dw + 4 * hl);
    f.q[1] = *(const uint4*)(row_dw + 4 * hl + 8);
    return f.v;
}

// Build a 32x16 f16 B-fragment from a row-major f32 weight matrix W[Krows][Nfull].
// Lane%16 = N column (n0 + lane%16); element j = K in A's table order within the 32-K step.
// Row at k==Krows (if brow != nullptr) is the bias vector: pairs with a constant
// 1.0 activation element so the WMMA accumulates the bias for free.
__device__ __forceinline__ v16h load_bfrag(const float* __restrict__ W,
                                           int Nfull, int Krows, int k, int n0,
                                           const float* __restrict__ brow = nullptr) {
    v16h b;
    const int kb = k & ~31, h = (k >> 4) & 1, n = n0 + (k & 15);
#pragma unroll
    for (int j = 0; j < 16; ++j) {
        const int kk = kb + ((j < 8) ? (8 * h + j) : (16 + 8 * h + j - 8));
        if (kk < Krows) b[j] = (_Float16)W[kk * Nfull + n];
        else if (brow != nullptr && kk == Krows) b[j] = (_Float16)brow[n];
        else b[j] = (_Float16)0.0f;
    }
    return b;
}

#define WMMA_F16(A, B, C) \
    __builtin_amdgcn_wmma_f32_16x16x32_f16(false, (A), false, (B), (short)0, (C), false, false)

__global__ __launch_bounds__(256, 1) void deepfm_fused_kernel(
    const int*   __restrict__ x_onehot,   // [B,2]
    const int*   __restrict__ multi_hot,  // [B,18]
    const float* __restrict__ emb_table,  // [38185,16]
    const float* __restrict__ fc_table,   // [38185,1]
    const float* __restrict__ W1, const float* __restrict__ b1,   // [48,64],[64]
    const float* __restrict__ W2, const float* __restrict__ b2,   // [64,32],[32]
    const float* __restrict__ W3, const float* __restrict__ b3,   // [32,16],[16]
    const float* __restrict__ W4, const float* __restrict__ b4,   // [16,1],[1]
    const float* __restrict__ bias,
    float* __restrict__ out, int total_chunks)
{
    // Per-wave private LDS slices; LDS ops are in-order within a wave -> no barriers.
    __shared__ __align__(16) _Float16 sh_h0[8][32 * 64];  // 32 samp x (48 feat + 1.0 + pad)
    __shared__ __align__(16) _Float16 sh_hb[8][16 * 64];  // mh staging / h1 / h2 / h3
    __shared__ float sh_fm[8][32];

    const int tid  = threadIdx.x;
    const int wave = tid >> 5;
    const int lane = tid & 31;
    const int hl   = lane >> 4;     // lane half
    const int nloc = lane & 15;

    // ---- Preload weight B-fragments into registers (loop-invariant) ----
    v16h B1c0[4], B1c1[4], B2c0[2], B2c1[2], B3f, Bg;
#pragma unroll
    for (int nt = 0; nt < 4; ++nt) {
        B1c0[nt] = load_bfrag(W1, 64, 48, lane,      nt * 16);
        B1c1[nt] = load_bfrag(W1, 64, 48, 32 + lane, nt * 16, b1);  // K=48 -> bias row
    }
#pragma unroll
    for (int nt = 0; nt < 2; ++nt) {
        B2c0[nt] = load_bfrag(W2, 32, 64, lane,      nt * 16);
        B2c1[nt] = load_bfrag(W2, 32, 64, 32 + lane, nt * 16);
    }
    B3f = load_bfrag(W3, 16, 32, lane, 0);
    // Genre table as constant B-fragment: eg = mh[16,18(pad32)] @ G[18,16]
    Bg  = load_bfrag(emb_table + EMB_OFFSET * EMB_D, 16, N_GENRES, lane, 0);

    float b2v[2];
#pragma unroll
    for (int nt = 0; nt < 2; ++nt) b2v[nt] = b2[nt * 16 + nloc];
    const float b3v = b3[nloc];
    float w4v[16];
#pragma unroll
    for (int n = 0; n < 16; ++n) w4v[n] = W4[n];
    float fcgv[N_GENRES];
#pragma unroll
    for (int g = 0; g < N_GENRES; ++g) fcgv[g] = fc_table[EMB_OFFSET + g];
    const float b4v = b4[0];
    const float bias0 = bias[0];

    // h0 pad (K=48..63) is chunk-invariant: K=48 = 1.0 (bias input), rest 0.
    {
        unsigned int* row32 = (unsigned int*)&sh_h0[wave][lane * 64];
        row32[24] = pack2(1.0f, 0.0f);
#pragma unroll
        for (int d = 1; d < 8; ++d) row32[24 + d] = 0u;
    }

    const v8f zero8 = {0.f, 0.f, 0.f, 0.f, 0.f, 0.f, 0.f, 0.f};
    const uint4 zero4 = {0u, 0u, 0u, 0u};

    // ---- Persistent grid-stride loop over 32-sample chunks ----
    for (int chunk = blockIdx.x * 8 + wave; chunk < total_chunks;
         chunk += NBLOCKS * 8) {
        const int s = chunk * 32 + lane;  // sample owned by this lane

        const int uid = x_onehot[2 * s + 0];
        const int iid = x_onehot[2 * s + 1];

        // multi-hot mask: 18 ints, 8B-aligned (72B stride) -> 9 x b64 loads
        int mh[18];
        {
            const int2* mp = (const int2*)(multi_hot + (size_t)s * N_GENRES);
#pragma unroll
            for (int q = 0; q < 9; ++q) {
                int2 v = mp[q];
                mh[2 * q + 0] = v.x;
                mh[2 * q + 1] = v.y;
            }
        }

        float eu[16], ei[16];
        {
            const float4* up = (const float4*)(emb_table + (size_t)uid * EMB_D);
            const float4* ip = (const float4*)(emb_table + (size_t)iid * EMB_D);
#pragma unroll
            for (int q = 0; q < 4; ++q) {
                float4 tu = up[q];
                eu[4 * q + 0] = tu.x; eu[4 * q + 1] = tu.y;
                eu[4 * q + 2] = tu.z; eu[4 * q + 3] = tu.w;
                float4 ti = ip[q];
                ei[4 * q + 0] = ti.x; ei[4 * q + 1] = ti.y;
                ei[4 * q + 2] = ti.z; ei[4 * q + 3] = ti.w;
            }
        }

        // Stage mh as f16 rows [32 samp][32 K] in sh_hb (K=18..31 zero)
        {
            uint4* row = (uint4*)&sh_hb[wave][lane * 32];
            uint4 q0, q1, q2;
            q0.x = pack2((float)mh[0],  (float)mh[1]);
            q0.y = pack2((float)mh[2],  (float)mh[3]);
            q0.z = pack2((float)mh[4],  (float)mh[5]);
            q0.w = pack2((float)mh[6],  (float)mh[7]);
            q1.x = pack2((float)mh[8],  (float)mh[9]);
            q1.y = pack2((float)mh[10], (float)mh[11]);
            q1.z = pack2((float)mh[12], (float)mh[13]);
            q1.w = pack2((float)mh[14], (float)mh[15]);
            q2.x = pack2((float)mh[16], (float)mh[17]);
            q2.y = 0u; q2.z = 0u; q2.w = 0u;
            row[0] = q0; row[1] = q1; row[2] = q2; row[3] = zero4;
        }

        // h0 row: eu | ei (dwords 0..15); eg filled by genre WMMA below
        {
            unsigned int* row32 = (unsigned int*)&sh_h0[wave][lane * 64];
#pragma unroll
            for (int d = 0; d < 8; ++d) row32[d]     = pack2(eu[2 * d], eu[2 * d + 1]);
#pragma unroll
            for (int d = 0; d < 8; ++d) row32[8 + d] = pack2(ei[2 * d], ei[2 * d + 1]);
        }

        // Genre embedding-bag via WMMA: one 16x16x32 per 16-sample subtile.
        // Result columns stored directly into h0 K=32..47.
#pragma unroll
        for (int t = 0; t < 2; ++t) {
            const unsigned int* mrow =
                (const unsigned int*)&sh_hb[wave][(t * 16 + nloc) * 32];
            const v16h am = load_afrag(mrow, hl);
            v8f accg = WMMA_F16(am, Bg, zero8);
#pragma unroll
            for (int r = 0; r < 8; ++r)
                sh_h0[wave][(t * 16 + r + 8 * hl) * 64 + 32 + nloc] =
                    (_Float16)accg[r];
        }

        // FM second-order + first-order (fp32): read back this lane's eg row.
        {
            const uint4* egq = (const uint4*)(&sh_h0[wave][lane * 64 + 32]);
            uint4 g0 = egq[0], g1 = egq[1];
            float eg[16];
            eg[0] = lo16(g0.x); eg[1] = hi16(g0.x);
            eg[2] = lo16(g0.y); eg[3] = hi16(g0.y);
            eg[4] = lo16(g0.z); eg[5] = hi16(g0.z);
            eg[6] = lo16(g0.w); eg[7] = hi16(g0.w);
            eg[8]  = lo16(g1.x); eg[9]  = hi16(g1.x);
            eg[10] = lo16(g1.y); eg[11] = hi16(g1.y);
            eg[12] = lo16(g1.z); eg[13] = hi16(g1.z);
            eg[14] = lo16(g1.w); eg[15] = hi16(g1.w);

            float fm2 = 0.0f;
#pragma unroll
            for (int d = 0; d < 16; ++d)
                fm2 += eu[d] * ei[d] + eg[d] * (eu[d] + ei[d]);
            float fcg = 0.0f;
#pragma unroll
            for (int g = 0; g < N_GENRES; ++g) fcg = fmaf((float)mh[g], fcgv[g], fcg);
            sh_fm[wave][lane] = bias0 + fc_table[uid] + fc_table[iid] + fcg + fm2;
        }

        // ---- MLP via WMMA: two 16-sample subtiles per wave ----
#pragma unroll
        for (int t = 0; t < 2; ++t) {
            const unsigned int* h0row =
                (const unsigned int*)&sh_h0[wave][(t * 16 + nloc) * 64];
            const v16h a0 = load_afrag(h0row,      hl);   // K = 0..31
            const v16h a1 = load_afrag(h0row + 16, hl);   // K = 32..47 | 1.0 | zeros

            // Layer 1: [16,48+bias]x[49,64] (bias folded into K=48)
            _Float16* h1 = &sh_hb[wave][0];               // [16][64]
#pragma unroll
            for (int nt = 0; nt < 4; ++nt) {
                v8f acc = WMMA_F16(a0, B1c0[nt], zero8);
                acc     = WMMA_F16(a1, B1c1[nt], acc);
#pragma unroll
                for (int r = 0; r < 8; ++r)
                    h1[(r + 8 * hl) * 64 + nt * 16 + nloc] =
                        (_Float16)fmaxf(acc[r], 0.0f);
            }

            // Layer 2: [16,64]x[64,32]
            const unsigned int* h1row = (const unsigned int*)&sh_hb[wave][nloc * 64];
            const v16h a20 = load_afrag(h1row,      hl);
            const v16h a21 = load_afrag(h1row + 16, hl);
            _Float16* h2 = &sh_hb[wave][0];               // reuse as [16][32]
#pragma unroll
            for (int nt = 0; nt < 2; ++nt) {
                v8f acc = WMMA_F16(a20, B2c0[nt], zero8);
                acc     = WMMA_F16(a21, B2c1[nt], acc);
#pragma unroll
                for (int r = 0; r < 8; ++r)
                    h2[(r + 8 * hl) * 32 + nt * 16 + nloc] =
                        (_Float16)fmaxf(acc[r] + b2v[nt], 0.0f);
            }

            // Layer 3: [16,32]x[32,16]
            const unsigned int* h2row = (const unsigned int*)&sh_hb[wave][nloc * 32];
            const v16h a3 = load_afrag(h2row, hl);
            v8f acc3 = WMMA_F16(a3, B3f, zero8);

            float* h3f = (float*)&sh_hb[wave][0];         // reuse as [16][16] f32
#pragma unroll
            for (int r = 0; r < 8; ++r)
                h3f[(r + 8 * hl) * 16 + nloc] = fmaxf(acc3[r] + b3v, 0.0f);

            // Layer 4 (16->1) + sigmoid, lanes 0..15 each finish one sample
            if (lane < 16) {
                float y = b4v;
#pragma unroll
                for (int n = 0; n < 16; ++n) y = fmaf(h3f[lane * 16 + n], w4v[n], y);
                const float z = sh_fm[wave][t * 16 + lane] + y;
                out[chunk * 32 + t * 16 + lane] = 1.0f / (1.0f + expf(-z));
            }
        }
    }
}

extern "C" void kernel_launch(void* const* d_in, const int* in_sizes, int n_in,
                              void* d_out, int out_size, void* d_ws, size_t ws_size,
                              hipStream_t stream) {
    const int*   x_onehot  = (const int*)d_in[0];
    const int*   multi_hot = (const int*)d_in[1];
    const float* emb_table = (const float*)d_in[2];
    const float* fc_table  = (const float*)d_in[3];
    const float* W1 = (const float*)d_in[4];
    const float* b1 = (const float*)d_in[5];
    const float* W2 = (const float*)d_in[6];
    const float* b2 = (const float*)d_in[7];
    const float* W3 = (const float*)d_in[8];
    const float* b3 = (const float*)d_in[9];
    const float* W4 = (const float*)d_in[10];
    const float* b4 = (const float*)d_in[11];
    const float* bias = (const float*)d_in[12];
    float* out = (float*)d_out;

    const int B = in_sizes[0] / 2;          // 524288
    const int total_chunks = B / 32;        // 16384 chunks of 32 samples

    deepfm_fused_kernel<<<NBLOCKS, 256, 0, stream>>>(
        x_onehot, multi_hot, emb_table, fc_table,
        W1, b1, W2, b2, W3, b3, W4, b4, bias, out, total_chunks);
}
